// LSTMLayer_18442589569677
// MI455X (gfx1250) — matrix operand order repaired and checked by reading, not compiled, not run on hardware
//
#include <hip/hip_runtime.h>

// LayerNorm-LSTM scan for MI455X (gfx1250, wave32, WMMA).
//  - Batch rows independent in the recurrence -> 4 workgroups x 16 rows,
//    whole 512-step scan in ONE kernel, all state in LDS (no grid sync).
//  - Input projection folded into recurrent GEMM: gates = [h | x_t] @ [Wh;Wx],
//    K=1024, via v_wmma_f32_16x16x32_bf16 (f32 accumulate).
//  - 512 threads/WG (16 waves, 4/SIMD) + register double-buffered B fragments
//    so weight loads from L2 overlap WMMA issue instead of serializing.
//  - Weights converted once to bf16 in d_ws (4 MB), L2-resident for the scan.

static constexpr int kT      = 512;   // SEQ
static constexpr int kIn     = 512;   // INPUT
static constexpr int kH      = 512;   // HIDDEN
static constexpr int kB      = 64;    // BATCH
static constexpr int kG4     = 2048;  // 4*H
static constexpr int kK      = 1024;  // H + INPUT (fused K)
static constexpr int kRows   = 16;    // batch rows per workgroup
static constexpr float kEps  = 1e-5f;

typedef __attribute__((ext_vector_type(16))) __bf16 v16bf;
typedef __attribute__((ext_vector_type(8)))  __bf16 v8bf;
typedef __attribute__((ext_vector_type(8)))  float  v8f;

__global__ void convert_f32_bf16(const float* __restrict__ src,
                                 __bf16* __restrict__ dst, int n) {
  int i = blockIdx.x * blockDim.x + threadIdx.x;
  int stride = gridDim.x * blockDim.x;
  for (; i < n; i += stride) dst[i] = (__bf16)src[i];
}

__device__ __forceinline__ float sigmoidf_(float x) {
  return 1.0f / (1.0f + __expf(-x));
}

// A fragment, 16-bit 16x32 WMMA layout: lanes<16 -> K {k0..+7, k0+16..+23},
// lanes>=16 -> K {k0+8..+15, k0+24..+31}
__device__ __forceinline__ v16bf load_a_frag(const __bf16* A_lds, int m, int hi, int k0) {
  union { v16bf v; v8bf h2[2]; } au;
  const __bf16* Ar = A_lds + m * kK + k0 + hi * 8;
  au.h2[0] = *(const v8bf*)(Ar);
  au.h2[1] = *(const v8bf*)(Ar + 16);
  return au.v;
}

// B fragment: column n = lane%16, K rows k0+8*(lane/16)+{0..7,16..23}; first 512 K-rows are Wh, rest Wx.
__device__ __forceinline__ const __bf16* b_row_ptr(const __bf16* Whb, const __bf16* Wxb,
                                                   int k0, int lane, int n_base) {
  return ((k0 < kH) ? (Whb + (size_t)(k0 + (lane >> 4) * 8) * kG4)
                    : (Wxb + (size_t)(k0 - kH + (lane >> 4) * 8) * kG4)) + n_base + (lane & 15);
}

__global__ __launch_bounds__(512, 1)
void lstm_scan_kernel(const float* __restrict__ x,
                      const float* __restrict__ bias,
                      const float* __restrict__ ln_c_w, const float* __restrict__ ln_c_b,
                      const float* __restrict__ ln_h_w, const float* __restrict__ ln_h_b,
                      const float* __restrict__ init_h, const float* __restrict__ init_c,
                      const __bf16* __restrict__ Whb, const __bf16* __restrict__ Wxb,
                      float* __restrict__ out) {
  // Dynamic LDS (~196 KB of the 320 KB WGP budget)
  extern __shared__ unsigned char smem[];
  __bf16* A_lds = (__bf16*)smem;                     // [16][1024] bf16: [h | x_t]   32 KB
  float*  gates = (float*)(smem + 32768);            // [16][2048] f32              128 KB
  float*  c_st  = (float*)(smem + 32768 + 131072);   // [16][512]  f32               32 KB
  float*  red_s = (float*)(smem + 196608);           // [16][32]
  float*  red_q = red_s + 512;                       // [16][32]
  float*  stats = red_q + 512;                       // [16][2] mean, inv-std

  const int tid  = threadIdx.x;
  const int wg   = blockIdx.x;                 // 0..3 -> batch rows wg*16..+15
  const int lane = tid & 31;
  const int wv   = tid >> 5;                   // wave 0..15
  const int r    = tid >> 5;                   // elementwise row 0..15 (32 thr/row)
  const int sub  = tid & 31;                   // 16 contiguous elems each
  const int m    = lane & 15;                  // WMMA A row / D column
  const int hi   = lane >> 4;                  // half-wave
  const int n_base = wv * 128;                 // this wave's 128 gate columns

  // ---- init h (bf16, A cols [0,512)) and c ----
  for (int k = 0; k < 16; ++k) {
    int j = sub * 16 + k;
    A_lds[r * kK + j] = (__bf16)init_h[j];
    c_st[r * kH + j]  = init_c[j];
  }

  const size_t out_hf = (size_t)kT * kB * kH;
  const size_t out_cf = out_hf + (size_t)kB * kH;

#pragma unroll 1
  for (int t = 0; t < kT; ++t) {
    // ---- stage x_t into A cols [512,1024) as bf16 ----
    for (int k = 0; k < 16; ++k) {
      int j = sub * 16 + k;
      A_lds[r * kK + kH + j] =
          (__bf16)x[((size_t)(wg * kRows + r) * kT + t) * kIn + j];
    }
    __syncthreads();

    // ---- GEMM: gates[16 x 2048] = A[16 x 1024] * [Wh;Wx][1024 x 2048] ----
    // Register double-buffered B fragments: chunk kc+1 loads overlap chunk kc WMMAs.
    v8f acc[8] = {};
    v16bf b0[8], b1[8];
    {
      const __bf16* bp = b_row_ptr(Whb, Wxb, 0, lane, n_base);
#pragma unroll
      for (int i = 0; i < 8; ++i) for (int s = 0; s < 16; ++s) b0[i][s] = bp[(size_t)(s + (s >> 3) * 8) * kG4 + i * 16];
    }
#pragma unroll 1
    for (int kc = 0; kc < 32; kc += 2) {
      v16bf a0 = load_a_frag(A_lds, m, hi, kc * 32);
      {
        const __bf16* bp = b_row_ptr(Whb, Wxb, (kc + 1) * 32, lane, n_base);
#pragma unroll
        for (int i = 0; i < 8; ++i) for (int s = 0; s < 16; ++s) b1[i][s] = bp[(size_t)(s + (s >> 3) * 8) * kG4 + i * 16];
      }
#pragma unroll
      for (int i = 0; i < 8; ++i)
        acc[i] = __builtin_amdgcn_wmma_f32_16x16x32_bf16(
            false, a0, false, b0[i], (short)0, acc[i], false, false);

      v16bf a1 = load_a_frag(A_lds, m, hi, (kc + 1) * 32);
      if (kc + 2 < 32) {
        const __bf16* bp = b_row_ptr(Whb, Wxb, (kc + 2) * 32, lane, n_base);
#pragma unroll
        for (int i = 0; i < 8; ++i) for (int s = 0; s < 16; ++s) b0[i][s] = bp[(size_t)(s + (s >> 3) * 8) * kG4 + i * 16];
      }
#pragma unroll
      for (int i = 0; i < 8; ++i)
        acc[i] = __builtin_amdgcn_wmma_f32_16x16x32_bf16(
            false, a1, false, b1[i], (short)0, acc[i], false, false);
    }
    // D layout: VGPR rr -> M = rr + 8*hi, N = lane%16
#pragma unroll
    for (int i = 0; i < 8; ++i) {
      const int ncol = n_base + i * 16 + m;
#pragma unroll
      for (int rr = 0; rr < 8; ++rr)
        gates[(hi * 8 + rr) * kG4 + ncol] = acc[i][rr];
    }
    __syncthreads();

    // ---- pass A: c_raw = sigmoid(f)*c + sigmoid(i)*tanh(g), partial sums ----
    float s = 0.f, q = 0.f;
    for (int k = 0; k < 16; ++k) {
      int j = sub * 16 + k;
      float gi = gates[r * kG4 + j]          + bias[j];
      float gf = gates[r * kG4 + kH + j]     + bias[kH + j];
      float gg = gates[r * kG4 + 3 * kH + j] + bias[3 * kH + j];
      float cr = sigmoidf_(gf) * c_st[r * kH + j] + sigmoidf_(gi) * tanhf(gg);
      c_st[r * kH + j] = cr;
      s += cr; q += cr * cr;
    }
    red_s[r * 32 + sub] = s; red_q[r * 32 + sub] = q;
    __syncthreads();
    if (sub == 0) {
      float ss = 0.f, qq = 0.f;
#pragma unroll
      for (int u = 0; u < 32; ++u) { ss += red_s[r * 32 + u]; qq += red_q[r * 32 + u]; }
      float mean = ss * (1.0f / kH);
      float var  = qq * (1.0f / kH) - mean * mean;
      stats[r * 2 + 0] = mean;
      stats[r * 2 + 1] = rsqrtf(var + kEps);
    }
    __syncthreads();

    // ---- pass B: c = LN(c_raw); h_raw = sigmoid(o)*tanh(c); partials ----
    {
      float mean = stats[r * 2 + 0], inv = stats[r * 2 + 1];
      s = 0.f; q = 0.f;
      for (int k = 0; k < 16; ++k) {
        int j = sub * 16 + k;
        float cl = (c_st[r * kH + j] - mean) * inv * ln_c_w[j] + ln_c_b[j];
        c_st[r * kH + j] = cl;
        float go = gates[r * kG4 + 2 * kH + j] + bias[2 * kH + j];
        float hr = sigmoidf_(go) * tanhf(cl);
        gates[r * kG4 + j] = hr;                 // reuse i-gate slot as h_raw
        s += hr; q += hr * hr;
      }
    }
    red_s[r * 32 + sub] = s; red_q[r * 32 + sub] = q;
    __syncthreads();
    if (sub == 0) {
      float ss = 0.f, qq = 0.f;
#pragma unroll
      for (int u = 0; u < 32; ++u) { ss += red_s[r * 32 + u]; qq += red_q[r * 32 + u]; }
      float mean = ss * (1.0f / kH);
      float var  = qq * (1.0f / kH) - mean * mean;
      stats[r * 2 + 0] = mean;
      stats[r * 2 + 1] = rsqrtf(var + kEps);
    }
    __syncthreads();

    // ---- pass C: h = LN(h_raw) -> outputs[t], bf16 h for next step ----
    {
      float mean = stats[r * 2 + 0], inv = stats[r * 2 + 1];
      for (int k = 0; k < 16; ++k) {
        int j = sub * 16 + k;
        float hl = (gates[r * kG4 + j] - mean) * inv * ln_h_w[j] + ln_h_b[j];
        A_lds[r * kK + j] = (__bf16)hl;
        out[((size_t)t * kB + (wg * kRows + r)) * kH + j] = hl;
        if (t == kT - 1)
          out[out_hf + (size_t)(wg * kRows + r) * kH + j] = hl;
      }
    }
    __syncthreads();  // protect gates/A_lds/stats reuse across iterations
  }

  // ---- final cell state ----
  for (int k = 0; k < 16; ++k) {
    int j = sub * 16 + k;
    out[out_cf + (size_t)(wg * kRows + r) * kH + j] = c_st[r * kH + j];
  }
}

extern "C" void kernel_launch(void* const* d_in, const int* in_sizes, int n_in,
                              void* d_out, int out_size, void* d_ws, size_t ws_size,
                              hipStream_t stream) {
  (void)in_sizes; (void)n_in; (void)out_size; (void)ws_size;
  const float* x    = (const float*)d_in[0];
  const float* Wx   = (const float*)d_in[1];
  const float* Wh   = (const float*)d_in[2];
  const float* bias = (const float*)d_in[3];
  const float* lncw = (const float*)d_in[4];
  const float* lncb = (const float*)d_in[5];
  const float* lnhw = (const float*)d_in[6];
  const float* lnhb = (const float*)d_in[7];
  const float* h0   = (const float*)d_in[8];
  const float* c0   = (const float*)d_in[9];
  float* out = (float*)d_out;

  // Workspace: bf16 weight copies (2 MB each), stay L2-resident for the scan.
  __bf16* Whb = (__bf16*)d_ws;
  __bf16* Wxb = Whb + (size_t)kH * kG4;

  const int nW = kH * kG4;  // 1,048,576 elements each
  convert_f32_bf16<<<(nW + 255) / 256, 256, 0, stream>>>(Wh, Whb, nW);
  convert_f32_bf16<<<(nW + 255) / 256, 256, 0, stream>>>(Wx, Wxb, nW);

  const size_t smem = 32768        // A_lds
                    + 131072       // gates
                    + 32768        // c_st
                    + 512 * 4 + 512 * 4 + 32 * 4;  // reductions + stats
  lstm_scan_kernel<<<kB / kRows, 512, smem, stream>>>(
      x, bias, lncw, lncb, lnhw, lnhb, h0, c0, Whb, Wxb, out);
}
